// CrossAttentionBlock_21560735826533
// MI455X (gfx1250) — matrix operand order repaired and checked by reading, not compiled, not run on hardware
//
#include <hip/hip_runtime.h>
#include <hip/hip_bf16.h>

#define DEV __device__ __forceinline__

typedef __attribute__((ext_vector_type(16))) __bf16         v16bf;
typedef __attribute__((ext_vector_type(2)))  __bf16         v2bf;
typedef __attribute__((ext_vector_type(16))) unsigned short v16u;
typedef __attribute__((ext_vector_type(8)))  unsigned short v8u;
typedef __attribute__((ext_vector_type(2)))  unsigned int   v2ui;
typedef __attribute__((ext_vector_type(8)))  float          v8f;
typedef __attribute__((ext_vector_type(4)))  float          v4f;
typedef __attribute__((ext_vector_type(2)))  float          v2f;

constexpr int   kB   = 32;
constexpr int   kC   = 512;
constexpr int   kHW  = 448;
constexpr int   kNH  = 8;
constexpr int   kDK  = 64;
constexpr int   kNQ  = kNH * kDK * 448;   // 229376 q outputs per batch
constexpr float kScale = 0.125f;          // 64^-0.5

// ---------------- helpers ----------------

DEV unsigned short f2bf(float f) {            // fp32 -> bf16 via native fptrunc
  __bf16 h = (__bf16)f;
  unsigned short u;
  __builtin_memcpy(&u, &h, 2);
  return u;
}

// Two floats -> packed 2xbf16 (v_cvt_pk_bf16_f32 via builtin or vector fptrunc).
DEV unsigned int f2bf2(float lo, float hi) {
#if __has_builtin(__builtin_amdgcn_cvt_pk_bf16_f32)
  auto p = __builtin_amdgcn_cvt_pk_bf16_f32(lo, hi);
  unsigned int u;
  __builtin_memcpy(&u, &p, 4);
  return u;
#else
  v2f f = {lo, hi};
  v2bf h = __builtin_convertvector(f, v2bf);
  unsigned int u;
  __builtin_memcpy(&u, &h, 4);
  return u;
#endif
}

// Build one 16-half fragment from two contiguous 16B LDS chunks (ds_load_b128 x2).
DEV v16bf frag_2x8(const unsigned short* p0, const unsigned short* p1) {
  v8u lo = *(const v8u*)p0;
  v8u hi = *(const v8u*)p1;
  v16u r = __builtin_shufflevector(lo, hi, 0, 1, 2, 3, 4, 5, 6, 7,
                                   8, 9, 10, 11, 12, 13, 14, 15);
  return __builtin_bit_cast(v16bf, r);
}

// A fragment 16(M)x32(K), source row-major [M][K] contiguous in K, sRow mult of 8.
// ISA: lane L<16 holds row M=L, K={kg..kg+7, kg+16..kg+23}, kg = (lane>>4)*8.
DEV v16bf fragA_c(const unsigned short* p, int sRow, int lane) {
  int row = lane & 15, kg = (lane >> 4) * 8;
  const unsigned short* q0 = p + row * sRow + kg;
  return frag_2x8(q0, q0 + 16);
}

// B fragment 32(K)x16(N), source stored K-TRANSPOSED: [N][K] contiguous in K.
// ISA: lane L<16 holds col N=L, K={kb..kb+7, kb+16..kb+23}, kb = (lane>>4)*8.
DEV v16bf fragBt(const unsigned short* p, int sN, int lane) {
  int col = lane & 15, kb = (lane >> 4) * 8;
  const unsigned short* q0 = p + col * sN + kb;
  return frag_2x8(q0, q0 + 16);
}

DEV v8f wmma_bf16(v16bf a, v16bf b, v8f c) {
  return __builtin_amdgcn_wmma_f32_16x16x32_bf16(false, a, false, b, (short)0, c,
                                                 false, false);
}

// ---------------- kernel 1: q = s @ W_q + b_q  (M=32, K=512, N=229376) ----------------

__global__ __launch_bounds__(256) void q_proj_kernel(
    const float* __restrict__ s, const float* __restrict__ Wq,
    const float* __restrict__ bq, unsigned short* __restrict__ q) {
  __shared__ unsigned short sl[32 * 40];      // s tile  [32m][32k]
  __shared__ unsigned short wl[256 * 40];     // Wq tile K-transposed [256n][32k]
  int t = threadIdx.x, lane = t & 31, w = t >> 5;
  int n0 = blockIdx.x * 256;
  int mtile = w & 1, nb = (w >> 1) * 64;      // wave -> 16 rows x 64 cols (4 tiles)

  v8f z = {0.f, 0.f, 0.f, 0.f, 0.f, 0.f, 0.f, 0.f};
  v8f acc[4];
#pragma unroll
  for (int j = 0; j < 4; ++j) acc[j] = z;

  for (int kk = 0; kk < 16; ++kk) {
    int k0 = kk * 32;
    {  // s tile: one float4 per thread, packed cvt + b64 store
      int row = t >> 3, c4 = t & 7;
      v4f f = *(const v4f*)(s + row * kC + k0 + c4 * 4);
      v2ui u = {f2bf2(f[0], f[1]), f2bf2(f[2], f[3])};
      *(v2ui*)(sl + row * 40 + c4 * 4) = u;
    }
    {  // W tile: float4 reads, transpose to [n][k]; strength-reduced addressing
      int r0 = t >> 6, c4 = t & 63;
      const float* wp = Wq + (size_t)(k0 + r0) * kNQ + n0 + c4 * 4;
      unsigned short* wd = wl + (c4 * 4) * 40 + r0;
#pragma unroll
      for (int rr = 0; rr < 8; ++rr) {
        v4f f = *(const v4f*)wp;
        wp += (size_t)4 * kNQ;
        unsigned int p0 = f2bf2(f[0], f[1]), p1 = f2bf2(f[2], f[3]);
        wd[0]   = (unsigned short)p0;
        wd[40]  = (unsigned short)(p0 >> 16);
        wd[80]  = (unsigned short)p1;
        wd[120] = (unsigned short)(p1 >> 16);
        wd += 4;
      }
    }
    if (kk < 15) {  // prefetch next k-tile of the dominant 470MB W_q stream
      int r = t >> 3, c = (t & 7) * 32;
      __builtin_prefetch(Wq + (size_t)(k0 + 32 + r) * kNQ + n0 + c, 0, 0);
    }
    __syncthreads();
    v16bf a = fragA_c(sl + mtile * 16 * 40, 40, lane);
#pragma unroll
    for (int j = 0; j < 4; ++j) {
      v16bf bb = fragBt(wl + (nb + j * 16) * 40, 40, lane);
      acc[j] = wmma_bf16(a, bb, acc[j]);
    }
    __syncthreads();
  }
  int col = lane & 15, rb = (lane >> 4) * 8;
#pragma unroll
  for (int j = 0; j < 4; ++j) {
    int n = n0 + nb + j * 16 + col;
    float bias = bq[n];
#pragma unroll
    for (int r = 0; r < 8; r += 2) {
      int m = mtile * 16 + rb + r;
      unsigned int p = f2bf2(acc[j][r] + bias, acc[j][r + 1] + bias);
      q[(size_t)m * kNQ + n]       = (unsigned short)p;
      q[(size_t)(m + 1) * kNQ + n] = (unsigned short)(p >> 16);
    }
  }
}

// ------------- kernel 2: kv = x_seq @ W_kv + b_kv, split into k/v bf16 -------------

__global__ __launch_bounds__(256) void kv_proj_kernel(
    const float* __restrict__ x, const float* __restrict__ Wkv,
    const float* __restrict__ bkv, unsigned short* __restrict__ ko,
    unsigned short* __restrict__ vo) {
  __shared__ unsigned short al[64 * 40];      // x tile  [64i][32k]
  __shared__ unsigned short bl[128 * 40];     // W tile K-transposed [128n][32k]
  int t = threadIdx.x, lane = t & 31, w = t >> 5;
  int blk = blockIdx.x;
  int b = blk / 56, rem = blk % 56;
  int mi = rem >> 3, nj = rem & 7;
  int i0 = mi * 64, n0 = nj * 128;
  int mtile = w & 3, grp = (w >> 2) * 64;

  v8f z = {0.f, 0.f, 0.f, 0.f, 0.f, 0.f, 0.f, 0.f};
  v8f acc[4];
#pragma unroll
  for (int j = 0; j < 4; ++j) acc[j] = z;

  for (int kk = 0; kk < 16; ++kk) {
    int k0 = kk * 32;
    {  // x[b][c][i]: float4 along i, 2 trips, strength-reduced
      int c0i = t >> 4, i4 = t & 15;
      const float* xp = x + ((size_t)b * kC + k0 + c0i) * kHW + i0 + i4 * 4;
      unsigned short* ad = al + (i4 * 4) * 40 + c0i;
#pragma unroll
      for (int cc = 0; cc < 2; ++cc) {
        v4f f = *(const v4f*)xp;
        xp += (size_t)16 * kHW;
        unsigned int p0 = f2bf2(f[0], f[1]), p1 = f2bf2(f[2], f[3]);
        ad[0]   = (unsigned short)p0;
        ad[40]  = (unsigned short)(p0 >> 16);
        ad[80]  = (unsigned short)p1;
        ad[120] = (unsigned short)(p1 >> 16);
        ad += 16;
      }
    }
    {  // W_kv: float4 along n, 4 trips, transpose
      int r0 = t >> 5, c4 = t & 31;
      const float* wp = Wkv + (size_t)(k0 + r0) * 1024 + n0 + c4 * 4;
      unsigned short* bd = bl + (c4 * 4) * 40 + r0;
#pragma unroll
      for (int rr = 0; rr < 4; ++rr) {
        v4f f = *(const v4f*)wp;
        wp += (size_t)8 * 1024;
        unsigned int p0 = f2bf2(f[0], f[1]), p1 = f2bf2(f[2], f[3]);
        bd[0]   = (unsigned short)p0;
        bd[40]  = (unsigned short)(p0 >> 16);
        bd[80]  = (unsigned short)p1;
        bd[120] = (unsigned short)(p1 >> 16);
        bd += 8;
      }
    }
    __syncthreads();
    v16bf a = fragA_c(al + mtile * 16 * 40, 40, lane);
#pragma unroll
    for (int j = 0; j < 4; ++j) {
      v16bf bb = fragBt(bl + (grp + j * 16) * 40, 40, lane);
      acc[j] = wmma_bf16(a, bb, acc[j]);
    }
    __syncthreads();
  }
  int col = lane & 15, rb = (lane >> 4) * 8;
#pragma unroll
  for (int j = 0; j < 4; ++j) {
    int n = n0 + grp + j * 16 + col;
    float bias = bkv[n];
    int hh = n >> 7, d = n & 127;
    unsigned short* dst = (d < 64) ? ko : vo;
    int dd = d & 63;
#pragma unroll
    for (int r = 0; r < 8; r += 2) {
      int m = i0 + mtile * 16 + rb + r;
      unsigned int p = f2bf2(acc[j][r] + bias, acc[j][r + 1] + bias);
      size_t o0 = (((size_t)b * kHW + m) * kNH + hh) * kDK + dd;
      dst[o0]                     = (unsigned short)p;
      dst[o0 + (size_t)kNH * kDK] = (unsigned short)(p >> 16);  // next m row
    }
  }
}

// ---------------- kernel 3: attention, one block per (b, head) ----------------
// S = q.k^T * scale; softmax over i (query axis) per column j -> independent
// per j-block; res += P @ v accumulated in registers across j-blocks.

constexpr int QS_ST = 72;   // ushort stride, q LDS [448][64]
constexpr int S_ST  = 32;   // float stride, S LDS [448][32] (column access: no conflicts)
constexpr int P_ST  = 40;   // ushort stride, P LDS [448][32]
constexpr int VT_ST = 40;   // ushort stride, v^T LDS [64][32]
constexpr size_t QS_B  = (size_t)448 * QS_ST * 2;   // 64512
constexpr size_t S_B   = (size_t)448 * S_ST * 4;    // 57344
constexpr size_t P_B   = (size_t)448 * P_ST * 2;    // 35840
constexpr size_t K_B   = (size_t)32 * QS_ST * 2;    // 4608
constexpr size_t VT_B  = (size_t)64 * VT_ST * 2;    // 5120
constexpr size_t ATTN_SMEM = QS_B + S_B + P_B + K_B + VT_B + (8 * 32 + 64) * 4;

__global__ __launch_bounds__(256) void attn_kernel(
    const unsigned short* __restrict__ qg, const unsigned short* __restrict__ kg,
    const unsigned short* __restrict__ vg, unsigned short* __restrict__ og) {
  extern __shared__ char smem[];
  unsigned short* qs    = (unsigned short*)smem;                 // [448][72]
  float*          S     = (float*)(smem + QS_B);                 // [448][32]
  unsigned short* Pb    = (unsigned short*)(smem + QS_B + S_B);  // [448][40]
  unsigned short* ksm   = Pb + 448 * P_ST;                       // [32][72]
  unsigned short* vsm_t = ksm + 32 * QS_ST;                      // v^T [64][40]
  float* redbuf = (float*)(vsm_t + 64 * VT_ST);                  // [8][32]
  float* colmax = redbuf + 8 * 32;                               // [32]
  float* colsum = colmax + 32;                                   // [32]

  int t = threadIdx.x, lane = t & 31, w = t >> 5;
  int b = blockIdx.x >> 3, h = blockIdx.x & 7;

  size_t qbase = (size_t)b * kNQ + h * 64;
  {  // q head: b128 both sides, 14 trips, strength-reduced
    int iq = t >> 3, d8 = t & 7;
    const unsigned short* qp = qg + qbase + (size_t)iq * 512 + d8 * 8;
    unsigned short* qd = qs + iq * QS_ST + d8 * 8;
#pragma unroll
    for (int ii = 0; ii < 14; ++ii) {
      *(v8u*)qd = *(const v8u*)qp;
      qp += 32 * 512;
      qd += 32 * QS_ST;
    }
  }
  v8f z = {0.f, 0.f, 0.f, 0.f, 0.f, 0.f, 0.f, 0.f};
  v8f res[14];
#pragma unroll
  for (int r = 0; r < 14; ++r) res[r] = z;
  __syncthreads();

  int col = t & 31, part = t >> 5;               // softmax mapping: 8 x 56 rows

  for (int jb = 0; jb < 14; ++jb) {
    int j0 = jb * 32;
    {  // k row-major (b128/b128), v transposed (b128 read, b16 scatter)
      int j = t >> 3, d8 = t & 7;
      size_t src = (((size_t)b * kHW + j0 + j) * kNH + h) * kDK + d8 * 8;
      *(v8u*)(ksm + j * QS_ST + d8 * 8) = *(const v8u*)(kg + src);
      v8u vv = *(const v8u*)(vg + src);
#pragma unroll
      for (int i = 0; i < 8; ++i) vsm_t[(d8 * 8 + i) * VT_ST + j] = vv[i];
    }
    __syncthreads();

    // ---- S block: 28 i-tiles x 2 j-tiles = 56 tiles, 7 per wave, K=64 ----
    int scl = lane & 15, srb = (lane >> 4) * 8;
#pragma unroll
    for (int tt = 0; tt < 7; ++tt) {
      int T = tt * 8 + w, it = T >> 1, jt = T & 1;
      v8f acc = z;
#pragma unroll
      for (int kq = 0; kq < 2; ++kq) {
        v16bf a  = fragA_c(qs + it * 16 * QS_ST + kq * 32, QS_ST, lane);
        v16bf bb = fragBt(ksm + jt * 16 * QS_ST + kq * 32, QS_ST, lane);
        acc = wmma_bf16(a, bb, acc);
      }
#pragma unroll
      for (int r = 0; r < 8; ++r)
        S[(it * 16 + srb + r) * S_ST + jt * 16 + scl] = acc[r] * kScale;
    }
    __syncthreads();

    // ---- column softmax over i = 0..447 ----
    int ib = part * 56;
    float m = -3.4e38f;
    for (int i = ib; i < ib + 56; ++i) m = fmaxf(m, S[i * S_ST + col]);
    redbuf[part * 32 + col] = m;
    __syncthreads();
    if (part == 0) {
      float mm = redbuf[col];
#pragma unroll
      for (int p = 1; p < 8; ++p) mm = fmaxf(mm, redbuf[p * 32 + col]);
      colmax[col] = mm;
    }
    __syncthreads();
    float mm = colmax[col];
    float sum = 0.f;
    for (int i = ib; i < ib + 56; ++i) {     // exp once: overwrite logits in place
      float e = __expf(S[i * S_ST + col] - mm);
      S[i * S_ST + col] = e;
      sum += e;
    }
    redbuf[part * 32 + col] = sum;
    __syncthreads();
    if (part == 0) {
      float ss = 0.f;
#pragma unroll
      for (int p = 0; p < 8; ++p) ss += redbuf[p * 32 + col];
      colsum[col] = ss;
    }
    __syncthreads();
    float inv = 1.f / colsum[col];
    for (int i = ib; i < ib + 56; i += 2) {  // scale + pack pairs
      unsigned int p = f2bf2(S[i * S_ST + col] * inv, S[(i + 1) * S_ST + col] * inv);
      Pb[i * P_ST + col]       = (unsigned short)p;
      Pb[(i + 1) * P_ST + col] = (unsigned short)(p >> 16);
    }
    __syncthreads();

    // ---- res += P(448x32) @ v(32x64): 112 tiles, 14 per wave, K=32 ----
#pragma unroll
    for (int tt = 0; tt < 14; ++tt) {
      int T = tt * 8 + w, it = T >> 2, dt = T & 3;
      v16bf a  = fragA_c(Pb + it * 16 * P_ST, P_ST, lane);
      v16bf bb = fragBt(vsm_t + dt * 16 * VT_ST, VT_ST, lane);
      res[tt] = wmma_bf16(a, bb, res[tt]);
    }
    __syncthreads();
  }

  int oc = lane & 15, orb = (lane >> 4) * 8;
#pragma unroll
  for (int tt = 0; tt < 14; ++tt) {
    int T = tt * 8 + w, it = T >> 2, dt = T & 3;
    int d = dt * 16 + oc;
#pragma unroll
    for (int r = 0; r < 8; r += 2) {
      int i = it * 16 + orb + r;
      unsigned int p = f2bf2(res[tt][r], res[tt][r + 1]);
      size_t o = ((size_t)b * kHW + i) * 512 + h * 64 + d;
      og[o]       = (unsigned short)p;
      og[o + 512] = (unsigned short)(p >> 16);   // next i row
    }
  }
}

// ------- kernel 4: out = (res @ W_o + b_o + x_seq)^T, stored [b][c][hw] -------
// Computed transposed (A = W_o^T) so stores are contiguous along hw.

__global__ __launch_bounds__(256) void out_proj_kernel(
    const unsigned short* __restrict__ ag, const float* __restrict__ Wo,
    const float* __restrict__ bo, const float* __restrict__ x,
    float* __restrict__ out) {
  __shared__ unsigned short wl[64 * 40];   // W_o^T tile [64c][32n]
  __shared__ unsigned short bl[64 * 40];   // attn tile [64i][32n] (=B K-transposed)
  int t = threadIdx.x, lane = t & 31, w = t >> 5;
  int blk = blockIdx.x;
  int b = blk / 56, rem = blk % 56;
  int cg = rem / 7, ig = rem % 7;
  int c0 = cg * 64, i0 = ig * 64;
  int ctile = w & 3, ihalf = w >> 2;       // 2 i-tiles per wave

  v8f z = {0.f, 0.f, 0.f, 0.f, 0.f, 0.f, 0.f, 0.f};
  v8f acc[2] = {z, z};

  for (int kk = 0; kk < 16; ++kk) {
    int n0 = kk * 32;
    {  // W_o: float4 along c, 2 trips, transpose
      int r0 = t >> 4, c4 = t & 15;
      const float* wp = Wo + (size_t)(n0 + r0) * kC + c0 + c4 * 4;
      unsigned short* wd = wl + (c4 * 4) * 40 + r0;
#pragma unroll
      for (int rr = 0; rr < 2; ++rr) {
        v4f f = *(const v4f*)wp;
        wp += (size_t)16 * kC;
        unsigned int p0 = f2bf2(f[0], f[1]), p1 = f2bf2(f[2], f[3]);
        wd[0]   = (unsigned short)p0;
        wd[40]  = (unsigned short)(p0 >> 16);
        wd[80]  = (unsigned short)p1;
        wd[120] = (unsigned short)(p1 >> 16);
        wd += 16;
      }
    }
    {  // attn tile: bf16, b128 both sides (already [i][n] = [N][K])
      int ii = t >> 2, r8 = t & 3;
      *(v8u*)(bl + ii * 40 + r8 * 8) =
          *(const v8u*)(ag + ((size_t)b * kHW + i0 + ii) * 512 + n0 + r8 * 8);
    }
    __syncthreads();
    v16bf a = fragA_c(wl + ctile * 16 * 40, 40, lane);
#pragma unroll
    for (int j = 0; j < 2; ++j) {
      int itile = ihalf * 2 + j;
      v16bf bb = fragBt(bl + itile * 16 * 40, 40, lane);
      acc[j] = wmma_bf16(a, bb, acc[j]);
    }
    __syncthreads();
  }
  int col = lane & 15, rb = (lane >> 4) * 8;
#pragma unroll
  for (int j = 0; j < 2; ++j) {
    int itile = ihalf * 2 + j;
#pragma unroll
    for (int r = 0; r < 8; ++r) {
      int c = c0 + ctile * 16 + rb + r;
      int i = i0 + itile * 16 + col;
      size_t o = ((size_t)b * kC + c) * kHW + i;
      out[o] = acc[j][r] + bo[c] + x[o];   // residual: x_seq[b,i,c] == x[b,c,i]
    }
  }
}

// ---------------- host launcher ----------------
// Workspace: q bf16 (14.68 MB) + k bf16 + v bf16 + attn bf16 = 58,720,256 B.

extern "C" void kernel_launch(void* const* d_in, const int* in_sizes, int n_in,
                              void* d_out, int out_size, void* d_ws, size_t ws_size,
                              hipStream_t stream) {
  const float* x    = (const float*)d_in[0];
  const float* s    = (const float*)d_in[1];
  const float* W_kv = (const float*)d_in[2];
  const float* b_kv = (const float*)d_in[3];
  const float* W_q  = (const float*)d_in[4];
  const float* b_q  = (const float*)d_in[5];
  const float* W_o  = (const float*)d_in[6];
  const float* b_o  = (const float*)d_in[7];
  float* out = (float*)d_out;

  unsigned short* qws = (unsigned short*)d_ws;
  unsigned short* kws = qws + (size_t)kB * kNQ;
  unsigned short* vws = kws + (size_t)kB * kHW * kC;
  unsigned short* aws = vws + (size_t)kB * kHW * kC;

  q_proj_kernel<<<kNQ / 256, 256, 0, stream>>>(s, W_q, b_q, qws);
  kv_proj_kernel<<<kB * 56, 256, 0, stream>>>(x, W_kv, b_kv, kws, vws);
  attn_kernel<<<kB * kNH, 256, ATTN_SMEM, stream>>>(qws, kws, vws, aws);
  out_proj_kernel<<<kB * 56, 256, 0, stream>>>(aws, W_o, b_o, x, out);
}
